// ForwardConvScatterNoLoops_5566277616216
// MI455X (gfx1250) — matrix operand order repaired and checked by reading, not compiled, not run on hardware
//
#include <hip/hip_runtime.h>

// CDNA5 / gfx1250 implicit-GEMM depthwise conv + scatter splat.
// D(16x16 f32) = A(16x32 f16 PSF taps) x B(32x16 f16 image patch), accumulated
// with v_wmma_f32_16x16x32_f16 over 33 filter rows x 2 K-chunks.
// A fragments: 2x ds_load_b128 (aligned). B fragments: 8x ds_load_b32 via a
// parity-duplicated, reversed row buffer. Single barrier per block.

typedef _Float16     v16h __attribute__((ext_vector_type(16)));
typedef _Float16     v8h  __attribute__((ext_vector_type(8)));
typedef float        v8f  __attribute__((ext_vector_type(8)));
typedef unsigned int v8u  __attribute__((ext_vector_type(8)));

#define Cc   8
#define Kp   8
#define Sp   33
#define Hh   512
#define Ww   512
#define Hf   544          // H + S - 1
#define Wf   544
#define CH   1024
#define CW   1024
#define ANCH 271          // anchor (511//2 = 255) + R (16)

__global__ __launch_bounds__(256) void zero_kernel(float* out, int n) {
  int i = blockIdx.x * 256 + threadIdx.x;
  if (i < n) out[i] = 0.0f;
}

__global__ __launch_bounds__(256) void finalize_kernel(float* out, int n) {
  int i = blockIdx.x * 256 + threadIdx.x;
  if (i < n) out[i] = fmaxf(out[i], 0.0f) * 0.125f;   // relu / C
}

__global__ __launch_bounds__(256) void conv_splat_kernel(
    const float* __restrict__ x,       // [B, C, 512, 512]
    const float* __restrict__ psf,     // [C, K, 33, 33]
    const float* __restrict__ coords,  // [C, K, 2] (cy, cx)
    float* __restrict__ out)           // [B, 1024, 1024] f32 accumulator
{
  // PSF taps f32->f16, rows zero-padded to 64 cols (128B, 16B aligned rows).
  __shared__ __attribute__((aligned(16))) _Float16 sPsf[Kp][Sp][64];
  // Reversed row slices: sR0[u][p] = x[i-u, jb+127-p]; sR1[u][q] = sR0[u][q+1]
  // (parity-shifted duplicate so any lane reads 8 aligned dwords).
  __shared__ __attribute__((aligned(16))) _Float16 sR0[Sp][192];
  __shared__ __attribute__((aligned(16))) _Float16 sR1[Sp][192];
  __shared__ int sY0[Kp], sX0[Kp];

  const int tid = threadIdx.x;
  const int jb  = blockIdx.x * 128;         // pixel-group base (5 groups)
  const int i   = blockIdx.y;               // output row in [0, 544)
  const int bz  = blockIdx.z;               // b*8 + c
  const int b   = bz >> 3;
  const int c   = bz & 7;

  // --- stage PSF (valid taps, then zero padding) ---
  for (int idx = tid; idx < Kp * Sp * Sp; idx += 256) {
    int k = idx / (Sp * Sp), r = idx % (Sp * Sp);
    sPsf[k][r / Sp][r % Sp] = (_Float16)psf[(c * Kp + k) * (Sp * Sp) + r];
  }
  for (int idx = tid; idx < Kp * Sp * 31; idx += 256) {
    int k = idx / (Sp * 31), r = idx % (Sp * 31);
    sPsf[k][r / 31][Sp + r % 31] = (_Float16)0.0f;
  }
  // --- stage all 33 reversed image rows, branch-free (clamp + select) ---
  const size_t xbase = ((size_t)b * Cc + c) * Hh * Ww;
  for (int idx = tid; idx < Sp * 192; idx += 256) {
    int ur = idx / 192, p = idx % 192;
    int row = i - ur, col = jb + 127 - p;
    int rowc = min(max(row, 0), Hh - 1), colc = min(max(col, 0), Ww - 1);
    float xv = x[xbase + (size_t)rowc * Ww + colc];
    _Float16 hv = ((row == rowc) && (col == colc)) ? (_Float16)xv : (_Float16)0.0f;
    sR0[ur][p] = hv;
    sR1[ur][p == 0 ? 191 : p - 1] = hv;   // sR1[q] = val(q+1); slot 191 unread
  }
  if (tid < Kp) {
    float cy = coords[(c * Kp + tid) * 2 + 0];
    float cx = coords[(c * Kp + tid) * 2 + 1];
    sY0[tid] = __float2int_rn(cy) - ANCH;   // jnp.round == RNE
    sX0[tid] = __float2int_rn(cx) - ANCH;
  }
  __syncthreads();

  const int lane = tid & 31;
  const int wv   = tid >> 5;                // wave id 0..7 -> 16-pixel group
  const int half = lane >> 4;
  const int nm   = lane & 15;               // A/D: psf row (M); B/D: pixel (N)

  v8f acc = {};

  for (int u = 0; u < Sp; ++u) {
    if ((i - u) < 0 || (i - u) >= Hh) continue;   // uniform (rows staged zero)
    const v8h* ap = (const v8h*)&sPsf[nm & 7][u][0];  // rows 8-15 duplicate 0-7
#pragma unroll
    for (int vb = 0; vb < 2; ++vb) {
      // A: lanes 0-15 K{0..15}, lanes 16-31 K{16..31} -> 2x b128.
      v8h lo = ap[4 * vb + 2 * half];
      v8h hi = ap[4 * vb + 2 * half + 1];
      v16h a = __builtin_shufflevector(lo, hi, 0, 1, 2, 3, 4, 5, 6, 7,
                                       8, 9, 10, 11, 12, 13, 14, 15);
      // B: lane needs 16 consecutive reversed-row f16 starting at p0;
      // parity-select the shifted copy so the base dword is aligned.
      int p0 = 127 - 16 * wv - nm + vb * 32 + 16 * half;
      const unsigned* bp = (const unsigned*)((p0 & 1) ? &sR1[u][p0 - 1]
                                                      : &sR0[u][p0]);
      v8u bw;
#pragma unroll
      for (int g = 0; g < 8; ++g) bw[g] = bp[g];
      v16h bm = __builtin_bit_cast(v16h, bw);

      acc = __builtin_amdgcn_wmma_f32_16x16x32_f16(
          false, a, false, bm, (short)0, acc, false, false);
    }
  }

  // D: lane(half,nm) VGPR r = D[M = r + 8*half, N = nm]; rows 8-15 duplicate
  // rows 0-7 (A used nm&7), so split the 8 psf stores across the two halves.
  int j = jb + 16 * wv + nm;
  if (j < Wf) {
#pragma unroll
    for (int r = 0; r < 8; ++r) {
      if (((r >> 2) & 1) == half) {       // half 0 -> psf 0-3, half 1 -> 4-7
        int Y = sY0[r] + i;
        int X = sX0[r] + j;
        if (Y >= 0 && Y < CH && X >= 0 && X < CW)
          atomicAdd(&out[(size_t)b * CH * CW + (size_t)Y * CW + X], acc[r]);
      }
    }
  }
}

extern "C" void kernel_launch(void* const* d_in, const int* in_sizes, int n_in,
                              void* d_out, int out_size, void* d_ws, size_t ws_size,
                              hipStream_t stream) {
  const float* x      = (const float*)d_in[0];
  const float* psf    = (const float*)d_in[1];
  const float* coords = (const float*)d_in[2];
  float* out = (float*)d_out;
  int n = out_size;                              // 2 * 1024 * 1024

  zero_kernel<<<(n + 255) / 256, 256, 0, stream>>>(out, n);

  dim3 grid(5 /*pixel groups*/, Hf /*rows*/, 2 * Cc /*b*c*/);
  conv_splat_kernel<<<grid, 256, 0, stream>>>(x, psf, coords, out);

  finalize_kernel<<<(n + 255) / 256, 256, 0, stream>>>(out, n);
}
